// TorchFuseAllTiled_43817256354170
// MI455X (gfx1250) — matrix operand order repaired and checked by reading, not compiled, not run on hardware
//
#include <hip/hip_runtime.h>
#include <hip/hip_bf16.h>
#include <math.h>

// ---------------- problem constants ----------------
#define TOKENS 2048
#define DMODEL 2048
#define VOCAB  50257

// ---------------- tiling ----------------
#define TM 128              // tokens per block
#define TN 128              // vocab cols per block
#define KB 32               // K chunk (one wmma k-step)
#define LDSSTR 40           // KB + 8 bf16 pad (stride in elements, 80 bytes)
#define NVT ((VOCAB + TN - 1) / TN)   // 393 vocab tiles

typedef __attribute__((ext_vector_type(16))) __bf16 v16bf;
typedef __attribute__((ext_vector_type(8)))  float  v8f;

union Frag { uint4 u[2]; v16bf v; };

// fp32 -> bf16 (RNE) without relying on __bf16 scalar arithmetic
__device__ __forceinline__ unsigned short f2bf(float f) {
    unsigned int u = __float_as_uint(f);
    u += 0x7FFFu + ((u >> 16) & 1u);
    return (unsigned short)(u >> 16);
}
__device__ __forceinline__ float bf2f(unsigned short b) {
    return __uint_as_float(((unsigned int)b) << 16);
}

// ============================================================
// Kernel 1: per (token-tile, vocab-tile) block:
//   bf16x3 WMMA GEMM of 128x128 logit tile over K=2048,
//   then per-row (max, sum exp) partials -> ws.
// ============================================================
__global__ __launch_bounds__(256) void ce_partial_kernel(
    const float* __restrict__ X, const float* __restrict__ W,
    float2* __restrict__ partial)
{
    __shared__ alignas(16) unsigned short Ah[TM * LDSSTR];
    __shared__ alignas(16) unsigned short Al[TM * LDSSTR];
    __shared__ alignas(16) unsigned short Bh[TN * LDSSTR];
    __shared__ alignas(16) unsigned short Bl[TN * LDSSTR];

    const int tt   = blockIdx.x;           // token tile (fast dim -> W-slice L2 reuse)
    const int vt   = blockIdx.y;           // vocab tile
    const int tid  = threadIdx.x;
    const int wave = tid >> 5;
    const int lane = tid & 31;
    const int row0 = tt * TM;
    const int col0 = vt * TN;

    v8f acc[8];
    for (int i = 0; i < 8; ++i)
        for (int e = 0; e < 8; ++e) acc[i][e] = 0.0f;

    for (int kb = 0; kb < DMODEL; kb += KB) {
        // ---- stage A tile (128 x 32 f32 -> hi/lo bf16) ----
        #pragma unroll
        for (int i = 0; i < 4; ++i) {
            int idx = (i * 256 + tid) * 4;       // element in 128x32 tile
            int r = idx >> 5, c = idx & 31;
            float4 a = *(const float4*)(X + (size_t)(row0 + r) * DMODEL + kb + c);
            float v[4] = {a.x, a.y, a.z, a.w};
            #pragma unroll
            for (int j = 0; j < 4; ++j) {
                unsigned short h = f2bf(v[j]);
                float rem = v[j] - bf2f(h);
                Ah[r * LDSSTR + c + j] = h;
                Al[r * LDSSTR + c + j] = f2bf(rem);
            }
        }
        // ---- stage B tile = W rows (vocab-guarded) ----
        #pragma unroll
        for (int i = 0; i < 4; ++i) {
            int idx = (i * 256 + tid) * 4;
            int r = idx >> 5, c = idx & 31;
            int n = col0 + r;
            float4 b = make_float4(0.f, 0.f, 0.f, 0.f);
            if (n < VOCAB) b = *(const float4*)(W + (size_t)n * DMODEL + kb + c);
            float v[4] = {b.x, b.y, b.z, b.w};
            #pragma unroll
            for (int j = 0; j < 4; ++j) {
                unsigned short h = f2bf(v[j]);
                float rem = v[j] - bf2f(h);
                Bh[r * LDSSTR + c + j] = h;
                Bl[r * LDSSTR + c + j] = f2bf(rem);
            }
        }
        __syncthreads();

        // ---- A fragment: 16x32 bf16, documented wave32 layout ----
        // lanes 0-15: row M=lane, K 0..7 in v0..3 / 16..23 in v4..7
        // lanes 16-31: same rows, K 8..15 / 24..31
        const int arow  = wave * 16 + (lane & 15);
        const int kA    = (lane >> 4) * 8;
        Frag fAh, fAl;
        fAh.u[0] = *(const uint4*)&Ah[arow * LDSSTR + kA];
        fAh.u[1] = *(const uint4*)&Ah[arow * LDSSTR + 16 + kA];
        fAl.u[0] = *(const uint4*)&Al[arow * LDSSTR + kA];
        fAl.u[1] = *(const uint4*)&Al[arow * LDSSTR + 16 + kA];

        // ---- B fragments: 32x16 bf16; B[k][n] = W[n][k] (contiguous in K) ----
        const int bcol = lane & 15;
        const int kB   = (lane >> 4) * 8;
        #pragma unroll
        for (int nt = 0; nt < 8; ++nt) {
            const int brow = nt * 16 + bcol;
            Frag fBh, fBl;
            fBh.u[0] = *(const uint4*)&Bh[brow * LDSSTR + kB];
            fBh.u[1] = *(const uint4*)&Bh[brow * LDSSTR + kB + 16];
            fBl.u[0] = *(const uint4*)&Bl[brow * LDSSTR + kB];
            fBl.u[1] = *(const uint4*)&Bl[brow * LDSSTR + kB + 16];

            // bf16x3: hi*hi + hi*lo + lo*hi, fp32 accumulate
            acc[nt] = __builtin_amdgcn_wmma_f32_16x16x32_bf16(
                false, fAh.v, false, fBh.v, (short)0, acc[nt], false, false);
            acc[nt] = __builtin_amdgcn_wmma_f32_16x16x32_bf16(
                false, fAh.v, false, fBl.v, (short)0, acc[nt], false, false);
            acc[nt] = __builtin_amdgcn_wmma_f32_16x16x32_bf16(
                false, fAl.v, false, fBh.v, (short)0, acc[nt], false, false);
        }
        __syncthreads();
    }

    // ---- per-row online-softmax partials over this 128-col slice ----
    // C layout: VGPR r, lanes 0-15 -> (M=r, N=lane); lanes 16-31 -> (M=r+8, N=lane-16)
    bool valid[8];
    #pragma unroll
    for (int nt = 0; nt < 8; ++nt)
        valid[nt] = (col0 + nt * 16 + (lane & 15)) < VOCAB;

    #pragma unroll
    for (int r = 0; r < 8; ++r) {
        float m = -INFINITY;
        #pragma unroll
        for (int nt = 0; nt < 8; ++nt)
            if (valid[nt]) m = fmaxf(m, acc[nt][r]);
        // butterfly max within 16-lane half (masks < 16 stay in-half)
        for (int msk = 1; msk < 16; msk <<= 1)
            m = fmaxf(m, __shfl_xor(m, msk, 32));

        float s = 0.f;
        #pragma unroll
        for (int nt = 0; nt < 8; ++nt)
            if (valid[nt]) s += __expf(acc[nt][r] - m);
        for (int msk = 1; msk < 16; msk <<= 1)
            s += __shfl_xor(s, msk, 32);

        if ((lane & 15) == 0) {
            int tok = row0 + wave * 16 + (lane >> 4) * 8 + r;
            float2 p; p.x = m; p.y = s;
            partial[(size_t)tok * NVT + vt] = p;   // [token][vtile] -> coalesced read in k2
        }
    }
}

// ============================================================
// Kernel 2: per token — merge 393 (m,s) partials (log-sum-exp
// merge) + target-row dot, write per-token loss.
// ============================================================
__global__ __launch_bounds__(256) void ce_reduce_kernel(
    const float* __restrict__ X, const float* __restrict__ W,
    const int* __restrict__ target,
    const float2* __restrict__ partial, float* __restrict__ token_loss)
{
    const int t   = blockIdx.x;
    const int tid = threadIdx.x;

    float M = -INFINITY, S = 0.f;
    for (int i = tid; i < NVT; i += 256) {
        float2 p = partial[(size_t)t * NVT + i];
        if (p.x > M) { S = S * __expf(M - p.x) + p.y; M = p.x; }
        else         { S += p.y * __expf(p.x - M); }
    }

    const int tgt = target[t];
    float dot = 0.f;
    for (int k = tid * 4; k < DMODEL; k += 1024) {
        float4 a = *(const float4*)(X + (size_t)t  * DMODEL + k);
        float4 w = *(const float4*)(W + (size_t)tgt * DMODEL + k);
        dot += a.x * w.x + a.y * w.y + a.z * w.z + a.w * w.w;
    }

    __shared__ float sM[256], sS[256], sD[256];
    sM[tid] = M; sS[tid] = S; sD[tid] = dot;
    __syncthreads();
    for (int off = 128; off > 0; off >>= 1) {
        if (tid < off) {
            float m1 = sM[tid], s1 = sS[tid];
            float m2 = sM[tid + off], s2 = sS[tid + off];
            float Mn = fmaxf(m1, m2);
            sM[tid] = Mn;
            sS[tid] = s1 * __expf(m1 - Mn) + s2 * __expf(m2 - Mn);
            sD[tid] += sD[tid + off];
        }
        __syncthreads();
    }
    if (tid == 0)
        token_loss[t] = -(sD[0] - sM[0] - logf(sS[0]));
}

// ============================================================
// Kernel 3: deterministic fixed-order sum of 2048 token losses.
// ============================================================
__global__ __launch_bounds__(256) void ce_final_sum_kernel(
    const float* __restrict__ token_loss, float* __restrict__ out)
{
    __shared__ float s[256];
    const int tid = threadIdx.x;
    float a = 0.f;
    for (int i = tid; i < TOKENS; i += 256) a += token_loss[i];
    s[tid] = a;
    __syncthreads();
    for (int off = 128; off > 0; off >>= 1) {
        if (tid < off) s[tid] += s[tid + off];
        __syncthreads();
    }
    if (tid == 0) out[0] = s[0];
}

// ============================================================
extern "C" void kernel_launch(void* const* d_in, const int* in_sizes, int n_in,
                              void* d_out, int out_size, void* d_ws, size_t ws_size,
                              hipStream_t stream) {
    const float* X   = (const float*)d_in[0];   // [2048, 2048] fp32
    const float* W   = (const float*)d_in[1];   // [50257, 2048] fp32
    const int*   tgt = (const int*)d_in[2];     // [2048] int
    float*       out = (float*)d_out;           // scalar loss

    float2* partial    = (float2*)d_ws;                         // [TOKENS][NVT]
    float*  token_loss = (float*)((char*)d_ws +
                          (size_t)TOKENS * NVT * sizeof(float2));

    dim3 g1(TOKENS / TM, NVT);   // token tiles fast => same W slice co-resident in L2
    ce_partial_kernel<<<g1, 256, 0, stream>>>(X, W, partial);
    ce_reduce_kernel<<<TOKENS, 256, 0, stream>>>(X, W, tgt, partial, token_loss);
    ce_final_sum_kernel<<<1, 256, 0, stream>>>(token_loss, out);
}
